// GlobalAttention_5291399708852
// MI455X (gfx1250) — matrix operand order repaired and checked by reading, not compiled, not run on hardware
//
#include <hip/hip_runtime.h>
#include <hip/hip_bf16.h>

typedef __attribute__((ext_vector_type(16))) __bf16 v16bf;
typedef __attribute__((ext_vector_type(8)))  __bf16 v8bf;
typedef __attribute__((ext_vector_type(2)))  __bf16 v2bf;
typedef __attribute__((ext_vector_type(8)))  float  v8f;

#define NTOK 53
#define HEADS 8
#define CDIM 256

__device__ __forceinline__ v8f zero8() {
    v8f z = {0.f, 0.f, 0.f, 0.f, 0.f, 0.f, 0.f, 0.f};
    return z;
}

// A-fragment: two 16-byte chunks (lane-half covers 16 consecutive K: hi*16..+15)
__device__ __forceinline__ v16bf ldA(const __bf16* p0, const __bf16* p1) {
    union { uint4 u[2]; v16bf v; } f;
    f.u[0] = *(const uint4*)p0;
    f.u[1] = *(const uint4*)p1;
    return f.v;
}

// B-fragment: contiguous 32 bytes from B^T row (lane-half covers 16 consecutive K)
__device__ __forceinline__ v16bf ldB(const __bf16* p) {
    union { uint4 u[2]; v16bf v; } f;
    f.u[0] = *(const uint4*)p;
    f.u[1] = *(const uint4*)(p + 8);
    return f.v;
}

__global__ void prep_weights_kernel(const float* __restrict__ wqkv,
                                    const float* __restrict__ wproj,
                                    __bf16* __restrict__ wqkv_bf,
                                    __bf16* __restrict__ wproj_bf) {
    int i = blockIdx.x * blockDim.x + threadIdx.x;
    if (i < 768 * 256) wqkv_bf[i] = (__bf16)wqkv[i];
    if (i < 256 * 256) wproj_bf[i] = (__bf16)wproj[i];
}

__global__ void prep_bias_kernel(const float* __restrict__ rel_table,
                                 const int* __restrict__ rel_index,
                                 float* __restrict__ bias) {
    int i = blockIdx.x * blockDim.x + threadIdx.x;   // [8][64][64]
    if (i >= 8 * 64 * 64) return;
    int h = i >> 12, r = (i >> 6) & 63, c = i & 63;
    float v = 0.f;
    if (r >= 4 && r < NTOK && c >= 4 && c < NTOK) {
        int idx = rel_index[(r - 4) * 49 + (c - 4)];
        v = rel_table[idx * 8 + h];
    }
    bias[i] = v;
}

__global__ __launch_bounds__(256)
void attn_kernel(const float* __restrict__ x,
                 const float* __restrict__ b_proj,
                 const __bf16* __restrict__ wqkv_bf,
                 const __bf16* __restrict__ wproj_bf,
                 const float* __restrict__ bias_ws,
                 float* __restrict__ out) {
    __shared__ __bf16 sh_x[NTOK * CDIM];   // 27136 B  x[b] in bf16
    __shared__ __bf16 sh_qkP[64 * 64];     //  8192 B  q[64x32] | k[64x32], reused as P[64x64]
    __shared__ float  sh_S[64 * 64];       // 16384 B  logits
    __shared__ __bf16 sh_vt[32 * 64];      //  4096 B  v transposed [hd][token]
    __shared__ __bf16 sh_o[64 * 32];       //  4096 B  per-head attention output
    __shared__ float  sh_inv[64];          //   256 B  1/rowsum

    const int b    = blockIdx.x;
    const int tid  = threadIdx.x;
    const int lane = tid & 31;
    const int wave = tid >> 5;
    const int lr   = lane & 15;
    const int hi   = lane >> 4;
    const float scale = 0.17677669529663689f;      // 32^-0.5

    // ---- load x[b] -> bf16 LDS (pairs: float2 load, packed bf16, b32 store) ----
    const float* xg = x + (size_t)b * (NTOK * CDIM);
    for (int i = tid * 2; i < NTOK * CDIM; i += 512) {
        float2 f = *(const float2*)(xg + i);
        v2bf p;
        p.x = (__bf16)f.x;
        p.y = (__bf16)f.y;
        *(v2bf*)&sh_x[i] = p;
    }
    __syncthreads();

    // projection accumulators: 8 (16x16) tiles per wave, live across the head loop
    v8f acc[8];
#pragma unroll
    for (int t = 0; t < 8; ++t) acc[t] = zero8();

    for (int h = 0; h < HEADS; ++h) {
        // ---- Phase A: q,k,v for head h = x @ Wslice^T, 24 tiles, K=256 ----
#pragma unroll
        for (int t = 0; t < 3; ++t) {
            int id = wave * 3 + t;          // 0..23
            int mat   = id >> 3;            // 0=q 1=k 2=v
            int rem   = id & 7;
            int mtile = rem >> 1;           // 0..3
            int ntile = rem & 1;            // 0..1
            int arow  = mtile * 16 + lr;
            int ar    = arow < NTOK ? arow : NTOK - 1;   // clamp: keeps values finite
            const __bf16* aBase = sh_x + ar * CDIM;
            const __bf16* wB =
                wqkv_bf + (size_t)(mat * 256 + h * 32 + ntile * 16 + lr) * 256;
            v8f c = zero8();
#pragma unroll
            for (int kk = 0; kk < 8; ++kk) {
                v16bf a  = ldA(aBase + kk * 32 + hi * 16, aBase + kk * 32 + 8 + hi * 16);
                v16bf bb = ldB(wB + kk * 32 + hi * 16);
                c = __builtin_amdgcn_wmma_f32_16x16x32_bf16(false, a, false, bb,
                                                            (short)0, c, false, false);
            }
            int n = ntile * 16 + lr;
            if (mat == 2) {
                // v: transposed store -> lane's 8 values contiguous: one b128 store
                v8bf pv;
#pragma unroll
                for (int i = 0; i < 8; ++i) pv[i] = (__bf16)c[i];
                *(v8bf*)&sh_vt[n * 64 + mtile * 16 + hi * 8] = pv;
            } else {
                float sc = (mat == 0) ? scale : 1.f;
                __bf16* dst = sh_qkP + (mat == 0 ? 0 : 2048);
#pragma unroll
                for (int i = 0; i < 8; ++i) {
                    int m = mtile * 16 + hi * 8 + i;
                    dst[m * 32 + n] = (__bf16)(c[i] * sc);
                }
            }
        }
        __syncthreads();

        // ---- Phase B: S = q @ k^T, 16 tiles, single K=32 step ----
#pragma unroll
        for (int t = 0; t < 2; ++t) {
            int id = wave * 2 + t;          // 0..15
            int mtile = id >> 2, ntile = id & 3;
            const __bf16* qB = sh_qkP + (mtile * 16 + lr) * 32;
            const __bf16* kB = sh_qkP + 2048 + (ntile * 16 + lr) * 32;
            v16bf a  = ldA(qB + hi * 16, qB + 8 + hi * 16);
            v16bf bb = ldB(kB + hi * 16);
            v8f c = zero8();
            c = __builtin_amdgcn_wmma_f32_16x16x32_bf16(false, a, false, bb,
                                                        (short)0, c, false, false);
#pragma unroll
            for (int i = 0; i < 8; ++i)
                sh_S[(mtile * 16 + hi * 8 + i) * 64 + ntile * 16 + lr] = c[i];
        }
        __syncthreads();

        // ---- Phase C: softmax rows (unnormalized exp -> P, keep 1/sum) ----
        if (tid < 64) {
            int r = tid;
            if (r < NTOK) {
                const float* bi = bias_ws + ((h * 64 + r) * 64);
                const float* sr = sh_S + r * 64;
                float mx = -1e30f;
                for (int j = 0; j < NTOK; ++j) {
                    float v = sr[j] + bi[j];
                    mx = v > mx ? v : mx;
                }
                float s = 0.f;
                for (int j = 0; j < 64; j += 2) {
                    float e0 = 0.f, e1 = 0.f;
                    if (j < NTOK)     { e0 = __expf(sr[j] + bi[j] - mx);         s += e0; }
                    if (j + 1 < NTOK) { e1 = __expf(sr[j + 1] + bi[j + 1] - mx); s += e1; }
                    v2bf p;
                    p.x = (__bf16)e0;
                    p.y = (__bf16)e1;
                    *(v2bf*)&sh_qkP[r * 64 + j] = p;
                }
                sh_inv[r] = 1.f / s;
            } else {
                v2bf z;
                z.x = (__bf16)0.f;
                z.y = (__bf16)0.f;
                for (int j = 0; j < 64; j += 2) *(v2bf*)&sh_qkP[r * 64 + j] = z;
                sh_inv[r] = 0.f;
            }
        }
        __syncthreads();

        // ---- Phase D: out_h = P @ v, 8 tiles, K=64 (2 steps) ----
        {
            int mtile = wave >> 1, ntile = wave & 1;
            v8f c = zero8();
#pragma unroll
            for (int kk = 0; kk < 2; ++kk) {
                const __bf16* pB = sh_qkP + (mtile * 16 + lr) * 64 + kk * 32;
                const __bf16* vB = sh_vt + (ntile * 16 + lr) * 64 + kk * 32;
                v16bf a  = ldA(pB + hi * 16, pB + 8 + hi * 16);
                v16bf bb = ldB(vB + hi * 16);
                c = __builtin_amdgcn_wmma_f32_16x16x32_bf16(false, a, false, bb,
                                                            (short)0, c, false, false);
            }
#pragma unroll
            for (int i = 0; i < 8; ++i) {
                int m = mtile * 16 + hi * 8 + i;
                sh_o[m * 32 + ntile * 16 + lr] = (__bf16)(c[i] * sh_inv[m]);
            }
        }
        __syncthreads();

        // ---- Phase E: accumulate projection: final += out_h @ w_proj[:, h*32:+32]^T ----
#pragma unroll
        for (int t = 0; t < 8; ++t) {
            int id = wave * 8 + t;          // 0..63
            int mtile = id >> 4, ntile = id & 15;
            const __bf16* oB = sh_o + (mtile * 16 + lr) * 32;
            const __bf16* wB = wproj_bf + (size_t)(ntile * 16 + lr) * 256 + h * 32;
            v16bf a  = ldA(oB + hi * 16, oB + 8 + hi * 16);
            v16bf bb = ldB(wB + hi * 16);
            acc[t] = __builtin_amdgcn_wmma_f32_16x16x32_bf16(false, a, false, bb,
                                                             (short)0, acc[t], false, false);
        }
        __syncthreads();
    }

    // ---- store: out[b] = final + b_proj ----
    float* og = out + (size_t)b * (NTOK * CDIM);
#pragma unroll
    for (int t = 0; t < 8; ++t) {
        int id = wave * 8 + t;
        int mtile = id >> 4, ntile = id & 15;
        int col = ntile * 16 + lr;
        float bp = b_proj[col];
#pragma unroll
        for (int i = 0; i < 8; ++i) {
            int row = mtile * 16 + hi * 8 + i;
            if (row < NTOK) og[row * CDIM + col] = acc[t][i] + bp;
        }
    }
}

extern "C" void kernel_launch(void* const* d_in, const int* in_sizes, int n_in,
                              void* d_out, int out_size, void* d_ws, size_t ws_size,
                              hipStream_t stream) {
    const float* x     = (const float*)d_in[0];
    const float* wqkv  = (const float*)d_in[1];
    const float* wproj = (const float*)d_in[2];
    const float* bproj = (const float*)d_in[3];
    const float* rtab  = (const float*)d_in[4];
    const int*   ridx  = (const int*)d_in[5];

    __bf16* ws_qkv  = (__bf16*)d_ws;                                      // 768*256 bf16
    __bf16* ws_proj = ws_qkv + 768 * 256;                                 // 256*256 bf16
    float*  ws_bias = (float*)((char*)d_ws + (768 * 256 + 256 * 256) * 2); // [8][64][64] f32

    prep_weights_kernel<<<768, 256, 0, stream>>>(wqkv, wproj, ws_qkv, ws_proj);
    prep_bias_kernel<<<128, 256, 0, stream>>>(rtab, ridx, ws_bias);
    attn_kernel<<<4096, 256, 0, stream>>>(x, bproj, ws_qkv, ws_proj, ws_bias,
                                          (float*)d_out);
}
